// RankedListLoss_3556232921179
// MI455X (gfx1250) — matrix operand order repaired and checked by reading, not compiled, not run on hardware
//
#include <hip/hip_runtime.h>
#include <stdint.h>

typedef __attribute__((ext_vector_type(16))) __bf16 v16bf;
typedef __attribute__((ext_vector_type(8)))  float v8f;

#define NROWS 8192
#define DDIM  256
#define MARGIN_C 0.4f
#define ALPHA_C  1.2f

__device__ __forceinline__ unsigned short f32_to_bf16_rne(float f) {
    union { float f; unsigned int u; } c; c.f = f;
    unsigned int u = c.u;
    unsigned int lsb = (u >> 16) & 1u;
    u += 0x7fffu + lsb;
    return (unsigned short)(u >> 16);
}
__device__ __forceinline__ float bf16_to_f32(unsigned short h) {
    union { unsigned int u; float f; } c; c.u = ((unsigned int)h) << 16;
    return c.f;
}

// ---------------------------------------------------------------------------
// Kernel 1: L2-normalize rows, emit bf16 hi/lo split, row sq, zero accumulators
// ---------------------------------------------------------------------------
__global__ void rll_prep_kernel(const float* __restrict__ x,
                                unsigned short* __restrict__ xhi,
                                unsigned short* __restrict__ xlo,
                                float* __restrict__ sq,
                                float* __restrict__ ap_sum, float* __restrict__ cnt,
                                float* __restrict__ an_num, float* __restrict__ an_den) {
    __shared__ float red[256];
    const int r = blockIdx.x;
    const int t = threadIdx.x;
    float v = x[(size_t)r * DDIM + t];
    red[t] = v * v;
    __syncthreads();
    for (int s = 128; s > 0; s >>= 1) {
        if (t < s) red[t] += red[t + s];
        __syncthreads();
    }
    const float ssum = red[0];
    const float inv = 1.0f / (sqrtf(ssum) + 1e-12f);
    const float xn = v * inv;
    const unsigned short hi = f32_to_bf16_rne(xn);
    const float lo = xn - bf16_to_f32(hi);
    xhi[(size_t)r * DDIM + t] = hi;
    xlo[(size_t)r * DDIM + t] = f32_to_bf16_rne(lo);
    if (t == 0) {
        sq[r] = ssum * inv * inv;       // ||xn||^2 as the reference computes it
        ap_sum[r] = 0.0f; cnt[r] = 0.0f; an_num[r] = 0.0f; an_den[r] = 0.0f;
    }
}

// ---------------------------------------------------------------------------
// Kernel 2: fused WMMA Gram tiles + distance + masked row reductions
// Workgroup = 256 threads (8 wave32), tile = 128x128; wave owns 32x64.
// G = Xhi*Xhi^T + Xhi*Xlo^T + Xlo*Xhi^T  (3xBF16 split, fp32 accumulate)
// ---------------------------------------------------------------------------
__global__ __launch_bounds__(256)
void rll_gemm_kernel(const unsigned short* __restrict__ xhi,
                     const unsigned short* __restrict__ xlo,
                     const float* __restrict__ sq,
                     const int* __restrict__ labels,
                     float* __restrict__ ap_sum, float* __restrict__ cnt,
                     float* __restrict__ an_num, float* __restrict__ an_den) {
    const int lane = threadIdx.x & 31;
    const int wave = threadIdx.x >> 5;
    const int half = lane >> 4;     // 0: lanes 0-15, 1: lanes 16-31
    const int lcol = lane & 15;

    const int wr = wave >> 1;       // 0..3 -> 32-row strip
    const int wc = wave & 1;        // 0..1 -> 64-col strip
    const int row0 = blockIdx.y * 128 + wr * 32;
    const int col0 = blockIdx.x * 128 + wc * 64;

    v8f acc[2][4];
#pragma unroll
    for (int rt = 0; rt < 2; ++rt)
#pragma unroll
        for (int ct = 0; ct < 4; ++ct)
            acc[rt][ct] = (v8f){0.f,0.f,0.f,0.f,0.f,0.f,0.f,0.f};

    union Frag { v16bf v; uint4 q[2]; };

    for (int k0 = 0; k0 < DDIM; k0 += 32) {
        Frag ahi[2], alo[2], bhi[4], blo[4];
        // A fragments (16x32 bf16): lane holds row = base+lcol;
        // half 0 -> K {0..7, 16..23}, half 1 -> K {8..15, 24..31}
#pragma unroll
        for (int rt = 0; rt < 2; ++rt) {
            const int row = row0 + rt * 16 + lcol;
            const size_t base = (size_t)row * DDIM + k0 + half * 8;
            ahi[rt].q[0] = *(const uint4*)(xhi + base);
            ahi[rt].q[1] = *(const uint4*)(xhi + base + 16);
            alo[rt].q[0] = *(const uint4*)(xlo + base);
            alo[rt].q[1] = *(const uint4*)(xlo + base + 16);
        }
        // B fragments (32x16 bf16): B[k][n] = X[col0+n][k]; lane n holds
        // half 0 -> K {0..7, 16..23}, half 1 -> K {8..15, 24..31}
#pragma unroll
        for (int ct = 0; ct < 4; ++ct) {
            const int col = col0 + ct * 16 + lcol;
            const size_t base = (size_t)col * DDIM + k0 + half * 8;
            bhi[ct].q[0] = *(const uint4*)(xhi + base);
            bhi[ct].q[1] = *(const uint4*)(xhi + base + 16);
            blo[ct].q[0] = *(const uint4*)(xlo + base);
            blo[ct].q[1] = *(const uint4*)(xlo + base + 16);
        }
#pragma unroll
        for (int rt = 0; rt < 2; ++rt)
#pragma unroll
            for (int ct = 0; ct < 4; ++ct) {
                acc[rt][ct] = __builtin_amdgcn_wmma_f32_16x16x32_bf16(
                    false, ahi[rt].v, false, bhi[ct].v, (short)0, acc[rt][ct], false, false);
                acc[rt][ct] = __builtin_amdgcn_wmma_f32_16x16x32_bf16(
                    false, ahi[rt].v, false, blo[ct].v, (short)0, acc[rt][ct], false, false);
                acc[rt][ct] = __builtin_amdgcn_wmma_f32_16x16x32_bf16(
                    false, alo[rt].v, false, bhi[ct].v, (short)0, acc[rt][ct], false, false);
            }
    }

    // ---- fused epilogue: dist + masked terms, reduce across columns ----
    int   ljv[4];
    float sjv[4];
#pragma unroll
    for (int ct = 0; ct < 4; ++ct) {
        const int col = col0 + ct * 16 + lcol;
        ljv[ct] = labels[col];
        sjv[ct] = sq[col];
    }

#pragma unroll
    for (int rt = 0; rt < 2; ++rt) {
#pragma unroll
        for (int v = 0; v < 8; ++v) {
            // C-layout: VGPR v, lanes 0-15 -> M=v, lanes 16-31 -> M=v+8
            const int row = row0 + rt * 16 + half * 8 + v;
            const int li = labels[row];
            const float si = sq[row];
            float ap = 0.f, cn = 0.f, nm = 0.f, dn = 0.f;
#pragma unroll
            for (int ct = 0; ct < 4; ++ct) {
                const int col = col0 + ct * 16 + lcol;
                const float g = acc[rt][ct][v];
                const float d2 = si + sjv[ct] - 2.0f * g;
                const float d = sqrtf(fmaxf(d2, 1e-12f));
                if (li == ljv[ct]) {
                    if (row != col) {
                        ap += fmaxf(d + (MARGIN_C - ALPHA_C), 0.0f);
                        cn += 1.0f;
                    }
                } else if (d < ALPHA_C) {
                    const float tlm = ALPHA_C - d;
                    const float w = __expf(tlm);   // TVAL = 1.0
                    nm += tlm * w;
                    dn += w;
                }
            }
            // reduce over the 16 column-lanes of this half (masks stay in-half)
#pragma unroll
            for (int m = 1; m < 16; m <<= 1) {
                ap += __shfl_xor(ap, m, 32);
                cn += __shfl_xor(cn, m, 32);
                nm += __shfl_xor(nm, m, 32);
                dn += __shfl_xor(dn, m, 32);
            }
            if (lcol == 0) {
                atomicAdd(&ap_sum[row], ap);
                atomicAdd(&cnt[row],    cn);
                atomicAdd(&an_num[row], nm);
                atomicAdd(&an_den[row], dn);
            }
        }
    }
}

// ---------------------------------------------------------------------------
// Kernel 3: per-row loss -> mean scalar
// ---------------------------------------------------------------------------
__global__ void rll_finalize_kernel(const float* __restrict__ ap_sum,
                                    const float* __restrict__ cnt,
                                    const float* __restrict__ an_num,
                                    const float* __restrict__ an_den,
                                    float* __restrict__ out) {
    __shared__ float red[256];
    const int t = threadIdx.x;
    float s = 0.f;
    for (int r = t; r < NROWS; r += 256) {
        const float lap = ap_sum[r] / (cnt[r] + 1e-5f);
        const float lan = an_num[r] / (an_den[r] + 1e-5f);
        s += lap + lan;
    }
    red[t] = s;
    __syncthreads();
    for (int k = 128; k > 0; k >>= 1) {
        if (t < k) red[t] += red[t + k];
        __syncthreads();
    }
    if (t == 0) out[0] = red[0] / (float)NROWS;
}

extern "C" void kernel_launch(void* const* d_in, const int* in_sizes, int n_in,
                              void* d_out, int out_size, void* d_ws, size_t ws_size,
                              hipStream_t stream) {
    const float* x      = (const float*)d_in[0];
    const int*   labels = (const int*)d_in[1];
    float* out = (float*)d_out;

    char* ws = (char*)d_ws;
    unsigned short* xhi = (unsigned short*)ws;                              // 4 MB
    unsigned short* xlo = (unsigned short*)(ws + (size_t)NROWS * DDIM * 2); // 4 MB
    float* sq     = (float*)(ws + (size_t)NROWS * DDIM * 4);
    float* ap_sum = sq + NROWS;
    float* cntv   = ap_sum + NROWS;
    float* an_num = cntv + NROWS;
    float* an_den = an_num + NROWS;

    rll_prep_kernel<<<NROWS, 256, 0, stream>>>(x, xhi, xlo, sq,
                                               ap_sum, cntv, an_num, an_den);
    dim3 grid(NROWS / 128, NROWS / 128);
    rll_gemm_kernel<<<grid, 256, 0, stream>>>(xhi, xlo, sq, labels,
                                              ap_sum, cntv, an_num, an_den);
    rll_finalize_kernel<<<1, 256, 0, stream>>>(ap_sum, cntv, an_num, an_den, out);
}
